// MultiHeadAttention_27625229648310
// MI455X (gfx1250) — matrix operand order repaired and checked by reading, not compiled, not run on hardware
//
#include <hip/hip_runtime.h>

// ---------------------------------------------------------------------------
// Causal MHA forward for MI455X (gfx1250, wave32, WMMA).
// All matmuls use v_wmma_f32_16x16x32_f16 (f32 accumulate).
// attn ([B,H,S,S] fp32, ~537MB) is written exactly once (bandwidth floor).
// K tiles in the attn/ctx pass are staged into LDS with the gfx1250 async
// DMA path (global_load_async_to_lds_b128 + s_wait_asynccnt).
// ---------------------------------------------------------------------------

typedef __attribute__((ext_vector_type(16))) _Float16 v16h;
typedef __attribute__((ext_vector_type(8)))  float    v8f;

#define D_IN  1024
#define HD    1024   // NUM_HEADS * D_HEAD
#define NH    16
#define DH    64
#define SEQ   2048
#define BATCH 2
#define ROWS  (BATCH * SEQ)   // 4096

#if __has_builtin(__builtin_amdgcn_global_load_async_to_lds_b128)
#define USE_ASYNC_K 1
#else
#define USE_ASYNC_K 0
#endif

// --- WMMA fragment helpers (contiguous k order (k = 16h + s), wave32) -----------

// A fragment: 16x32 f16 tile of row-major A (ld in halves).
// lane<16 : row=lane,    halves 0..15 = K 0..15
// lane>=16: row=lane-16, halves 0..15 = K 16..31
__device__ __forceinline__ v16h load_a16(const _Float16* A, int lda,
                                         int m0, int kk, int lane) {
  const _Float16* p = A + (size_t)(m0 + (lane & 15)) * lda + kk + ((lane >> 4) << 4);
  v16h r;
  ((uint4*)&r)[0] = *(const uint4*)(p);        // 8 halves, 16B
  ((uint4*)&r)[1] = *(const uint4*)(p + 8);    // 8 halves, 16B
  return r;
}

// B fragment: 32x16 f16 tile, sourced from a pre-transposed matrix BT[n][k]
// (ld in halves).  lane<16: col n=lane, K 0..15; lane>=16: col=lane-16, K 16..31.
__device__ __forceinline__ v16h load_bt16(const _Float16* BT, int ldb,
                                          int n0, int kk, int lane) {
  const _Float16* p = BT + (size_t)(n0 + (lane & 15)) * ldb + kk + ((lane >> 4) << 4);
  v16h r;
  ((uint4*)&r)[0] = *(const uint4*)(p);        // K .. K+7
  ((uint4*)&r)[1] = *(const uint4*)(p + 8);    // K+8 .. K+15
  return r;
}

__device__ __forceinline__ v8f wmma16(v16h a, v16h b, v8f c) {
  return __builtin_amdgcn_wmma_f32_16x16x32_f16(false, a, false, b,
                                                (short)0, c, false, false);
}

// per-lane 16B async DMA global -> LDS (gfx1250 ASYNCcnt path)
__device__ __forceinline__ void async_b128(const _Float16* g, _Float16* l) {
#if USE_ASYNC_K
  typedef int v4i_t __attribute__((vector_size(16)));
  typedef v4i_t __attribute__((address_space(3))) lds_v4i_t;
  __builtin_amdgcn_global_load_async_to_lds_b128(
      (v4i_t*)g,
      (lds_v4i_t*)(unsigned int)(unsigned long long)l,
      0, 0);
#else
  (void)g; (void)l;
#endif
}

// --- conversion / transpose kernels ----------------------------------------

__global__ void k_f32_to_f16(const float* __restrict__ src,
                             _Float16* __restrict__ dst, int n) {
  int i = blockIdx.x * blockDim.x + threadIdx.x;
  if (i < n) dst[i] = (_Float16)src[i];
}

// W: [K, N=2^logN] f32 row-major  ->  WT: [N, K] f16 row-major
__global__ void k_transpose_to_f16(const float* __restrict__ W,
                                   _Float16* __restrict__ WT, int K, int logN) {
  int i = blockIdx.x * blockDim.x + threadIdx.x;
  int N = 1 << logN;
  if (i < (K << logN)) {
    int k = i >> logN, n = i & (N - 1);
    WT[(size_t)n * K + k] = (_Float16)W[i];
  }
}

// --- QKV projection: 32x64 output per wave (2x4 register blocking) ----------
// which=0 -> Qh[B,H,S,DH]; which=1 -> Kh[B,H,S,DH]; which=2 -> Vt[B,H,DH,S]

__global__ void k_qkv(const _Float16* __restrict__ xh,
                      const _Float16* __restrict__ WqT,
                      const _Float16* __restrict__ WkT,
                      const _Float16* __restrict__ WvT,
                      _Float16* __restrict__ Qh,
                      _Float16* __restrict__ Kh,
                      _Float16* __restrict__ Vt) {
  const int lane = threadIdx.x & 31;
  const int MB = ROWS / 32, NB = HD / 64;       // 128 x 16
  int which = blockIdx.x / (MB * NB);
  int rem   = blockIdx.x % (MB * NB);
  int m0 = (rem / NB) * 32, n0 = (rem % NB) * 64;
  const _Float16* BT = (which == 0) ? WqT : (which == 1) ? WkT : WvT;

  v8f acc[2][4] = {};
  for (int kk = 0; kk < D_IN; kk += 32) {
    v16h a0 = load_a16(xh, D_IN, m0,      kk, lane);
    v16h a1 = load_a16(xh, D_IN, m0 + 16, kk, lane);
    for (int j = 0; j < 4; ++j) {
      v16h b = load_bt16(BT, D_IN, n0 + 16 * j, kk, lane);
      acc[0][j] = wmma16(a0, b, acc[0][j]);
      acc[1][j] = wmma16(a1, b, acc[1][j]);
    }
  }

  const int col = lane & 15, rbase = (lane >> 4) << 3;
  for (int i = 0; i < 2; ++i)
    for (int j = 0; j < 4; ++j)
      for (int r = 0; r < 8; ++r) {
        int m = m0 + 16 * i + rbase + r;    // b*SEQ + s
        int n = n0 + 16 * j + col;          // h*DH + d
        int b = m / SEQ, s = m % SEQ;       // power-of-two: shift/mask
        int h = n / DH,  d = n % DH;
        _Float16 v = (_Float16)acc[i][j][r];
        if (which == 0)      Qh[(((size_t)b * NH + h) * SEQ + s) * DH + d] = v;
        else if (which == 1) Kh[(((size_t)b * NH + h) * SEQ + s) * DH + d] = v;
        else                 Vt[(((size_t)b * NH + h) * DH + d) * SEQ + s] = v;
      }
}

// --- pass 1: online softmax stats (row max M, row sum L) --------------------
// one wave per (b,h, 16-row q-tile); scores recomputed, never stored.

__global__ void k_softmax_stats(const _Float16* __restrict__ Qh,
                                const _Float16* __restrict__ Kh,
                                float* __restrict__ M, float* __restrict__ L) {
  const int lane = threadIdx.x & 31;
  const int QT = SEQ / 16;
  int qt = blockIdx.x % QT;
  int bh = blockIdx.x / QT;
  const _Float16* Q = Qh + (size_t)bh * SEQ * DH;
  const _Float16* K = Kh + (size_t)bh * SEQ * DH;
  int q0 = qt * 16;
  const int col = lane & 15, rbase = (lane >> 4) << 3;

  float m[8], s[8];
  for (int r = 0; r < 8; ++r) { m[r] = -__builtin_inff(); s[r] = 0.f; }

  v16h aq0 = load_a16(Q, DH, q0, 0, lane);
  v16h aq1 = load_a16(Q, DH, q0, 32, lane);

  for (int kt = 0; kt <= qt; ++kt) {
    int k0 = kt * 16;
    if (kt + 1 <= qt)   // pull next 2KB K tile toward the WGP (global_prefetch)
      __builtin_prefetch(K + (size_t)(k0 + 16) * DH + lane * 8, 0, 3);
    v8f sc = {};
    sc = wmma16(aq0, load_bt16(K, DH, k0, 0, lane), sc);
    sc = wmma16(aq1, load_bt16(K, DH, k0, 32, lane), sc);
    bool diag = (kt == qt);
    for (int r = 0; r < 8; ++r) {
      int q = q0 + rbase + r, k = k0 + col;
      // mask before scale (matches reference; -inf survives /8)
      float v = (diag && k > q) ? -__builtin_inff() : sc[r] * 0.125f;
      // row lives in one 16-lane half of the wave: xor masks 1..8 stay in-half
      float tm = v;
      tm = fmaxf(tm, __shfl_xor(tm, 1, 32));
      tm = fmaxf(tm, __shfl_xor(tm, 2, 32));
      tm = fmaxf(tm, __shfl_xor(tm, 4, 32));
      tm = fmaxf(tm, __shfl_xor(tm, 8, 32));
      float nm = fmaxf(m[r], tm);
      float e  = __expf(v - nm);
      e += __shfl_xor(e, 1, 32);
      e += __shfl_xor(e, 2, 32);
      e += __shfl_xor(e, 4, 32);
      e += __shfl_xor(e, 8, 32);
      s[r] = s[r] * __expf(m[r] - nm) + e;
      m[r] = nm;
    }
  }
  if (col == 0) {  // lanes 0 and 16 write rows 0..7 / 8..15
    for (int r = 0; r < 8; ++r) {
      int q = q0 + rbase + r;
      M[(size_t)bh * SEQ + q] = m[r];
      L[(size_t)bh * SEQ + q] = s[r];
    }
  }
}

// --- pass 2: write normalized attn (single write) + fused ctx = P @ V -------
// K tiles double-buffered in LDS via async DMA; P goes through a 16x32 LDS
// tile to convert D-layout -> A-layout for the P@V WMMA.

__global__ void k_attn_ctx(const _Float16* __restrict__ Qh,
                           const _Float16* __restrict__ Kh,
                           const _Float16* __restrict__ Vt,
                           const float* __restrict__ M,
                           const float* __restrict__ L,
                           float* __restrict__ attn,
                           _Float16* __restrict__ ctxh) {
  __shared__ __attribute__((aligned(16))) _Float16 ptile[16 * 32];
#if USE_ASYNC_K
  __shared__ __attribute__((aligned(16))) _Float16 kbuf[2][16 * DH];
#endif
  const int lane = threadIdx.x & 31;
  const int QT = SEQ / 16;
  int qt = blockIdx.x % QT;
  int bh = blockIdx.x / QT;
  int b = bh / NH, h = bh % NH;
  const _Float16* Q = Qh + (size_t)bh * SEQ * DH;
  const _Float16* K = Kh + (size_t)bh * SEQ * DH;
  const _Float16* V = Vt + (size_t)bh * DH * SEQ;   // [DH][SEQ]
  float* A = attn + (size_t)bh * SEQ * SEQ;
  int q0 = qt * 16;
  const int col = lane & 15, rbase = (lane >> 4) << 3;

  // zero-fill strictly-above-diagonal region (softmax(-inf) == 0 exactly)
  // division-free: row-outer loop, lane-strided float4 columns
  {
    int kstart = (qt + 1) * 16;
    if (kstart < SEQ) {
      float4 z = make_float4(0.f, 0.f, 0.f, 0.f);
      for (int r = 0; r < 16; ++r) {
        float* row = A + (size_t)(q0 + r) * SEQ;
        for (int c = kstart + lane * 4; c < SEQ; c += 128)
          *(float4*)(row + c) = z;
      }
    }
  }

  float mr[8], rl[8];
  for (int r = 0; r < 8; ++r) {
    int q = q0 + rbase + r;
    mr[r] = M[(size_t)bh * SEQ + q];
    rl[r] = 1.0f / L[(size_t)bh * SEQ + q];
  }

  v16h aq0 = load_a16(Q, DH, q0, 0, lane);
  v16h aq1 = load_a16(Q, DH, q0, 32, lane);
  v8f acc[4] = {};

#if USE_ASYNC_K
  // each 16-key tile of Kh is one contiguous 2KB block: 4 lane-strided b128 DMAs
  auto stage_k = [&](int kt, int buf) {
    const _Float16* src = K + (size_t)kt * 16 * DH;
    _Float16* dst = &kbuf[buf][0];
    int c = lane * 8;                       // 16B per lane
    async_b128(src + c,       dst + c);
    async_b128(src + c + 256, dst + c + 256);
    async_b128(src + c + 512, dst + c + 512);
    async_b128(src + c + 768, dst + c + 768);
  };
  stage_k(0, 0);
#endif

  for (int kt = 0; kt <= qt; kt += 2) {       // 32 keys per step
    for (int t = 0; t < 2; ++t) {
      int ktt = kt + t;
      if (ktt <= qt) {
        int k0 = ktt * 16;
        v8f sc = {};
#if USE_ASYNC_K
        bool staged = (ktt + 1 <= qt);
        if (staged) {
          // previous reads of the target buffer retired (their data fed wmma)
          asm volatile("s_wait_dscnt 0x0" ::: "memory");
          stage_k(ktt + 1, (ktt + 1) & 1);
          asm volatile("s_wait_asynccnt 0x4" ::: "memory");  // current tile in
        } else {
          asm volatile("s_wait_asynccnt 0x0" ::: "memory");
        }
        {
          const _Float16* kb = &kbuf[ktt & 1][(lane & 15) * DH + ((lane >> 4) << 4)];
          v16h b0, b1;
          ((uint4*)&b0)[0] = *(const uint4*)(kb);
          ((uint4*)&b0)[1] = *(const uint4*)(kb + 8);
          ((uint4*)&b1)[0] = *(const uint4*)(kb + 32);
          ((uint4*)&b1)[1] = *(const uint4*)(kb + 40);
          sc = wmma16(aq0, b0, sc);
          sc = wmma16(aq1, b1, sc);
        }
#else
        sc = wmma16(aq0, load_bt16(K, DH, k0, 0, lane), sc);
        sc = wmma16(aq1, load_bt16(K, DH, k0, 32, lane), sc);
#endif
        bool diag = (ktt == qt);
        for (int r = 0; r < 8; ++r) {
          int q = q0 + rbase + r, k = k0 + col;
          float p = (diag && k > q) ? 0.f
                                    : __expf(sc[r] * 0.125f - mr[r]) * rl[r];
          A[(size_t)q * SEQ + k] = p;                         // the one write
          ptile[(rbase + r) * 32 + t * 16 + col] = (_Float16)p;
        }
      } else {
        for (int r = 0; r < 8; ++r)
          ptile[(rbase + r) * 32 + t * 16 + col] = (_Float16)0.f;
      }
    }
    // in-wave LDS store->load ordering before reading P as A-fragment
    asm volatile("s_wait_dscnt 0x0" ::: "memory");
    v16h ap;
    {
      const _Float16* p = &ptile[(lane & 15) * 32 + ((lane >> 4) << 4)];
      ((uint4*)&ap)[0] = *(const uint4*)(p);
      ((uint4*)&ap)[1] = *(const uint4*)(p + 8);
    }
    int kv0 = kt * 16;
    for (int nt = 0; nt < 4; ++nt)
      acc[nt] = wmma16(ap, load_bt16(V, SEQ, nt * 16, kv0, lane), acc[nt]);
  }

  // ctx -> f16, layout [B,S, H*DH] for the out-projection GEMM
  for (int nt = 0; nt < 4; ++nt)
    for (int r = 0; r < 8; ++r) {
      int q = q0 + rbase + r;
      int d = nt * 16 + col;
      ctxh[((size_t)b * SEQ + q) * HD + h * DH + d] = (_Float16)acc[nt][r];
    }
}

// --- out = ctx @ Wo ---------------------------------------------------------

__global__ void k_outproj(const _Float16* __restrict__ ctxh,
                          const _Float16* __restrict__ WoT,
                          float* __restrict__ out) {
  const int lane = threadIdx.x & 31;
  int m0 = (blockIdx.x >> 2) * 16, n0 = (blockIdx.x & 3) * 16;
  v8f acc = {};
  for (int kk = 0; kk < HD; kk += 32)
    acc = wmma16(load_a16(ctxh, HD, m0, kk, lane),
                 load_bt16(WoT, HD, n0, kk, lane), acc);
  const int col = lane & 15, rbase = (lane >> 4) << 3;
  for (int r = 0; r < 8; ++r)
    out[(size_t)(m0 + rbase + r) * DH + n0 + col] = acc[r];
}

// ---------------------------------------------------------------------------

extern "C" void kernel_launch(void* const* d_in, const int* in_sizes, int n_in,
                              void* d_out, int out_size, void* d_ws, size_t ws_size,
                              hipStream_t stream) {
  (void)in_sizes; (void)n_in; (void)out_size; (void)ws_size;
  const float* x  = (const float*)d_in[0];
  const float* Wq = (const float*)d_in[1];
  const float* Wk = (const float*)d_in[2];
  const float* Wv = (const float*)d_in[3];
  const float* Wo = (const float*)d_in[4];

  float* out  = (float*)d_out;                       // [B,S,DH]
  float* attn = out + (size_t)BATCH * SEQ * DH;      // [B,H,S,S]

  char* ws = (char*)d_ws;
  size_t off = 0;
  auto alloc = [&](size_t bytes) {
    void* p = ws + off;
    off = (off + bytes + 255) & ~(size_t)255;
    return p;
  };
  _Float16* xh   = (_Float16*)alloc((size_t)ROWS * D_IN * 2);
  _Float16* WqT  = (_Float16*)alloc((size_t)D_IN * HD * 2);
  _Float16* WkT  = (_Float16*)alloc((size_t)D_IN * HD * 2);
  _Float16* WvT  = (_Float16*)alloc((size_t)D_IN * HD * 2);
  _Float16* WoT  = (_Float16*)alloc((size_t)HD * DH * 2);
  _Float16* Qh   = (_Float16*)alloc((size_t)BATCH * NH * SEQ * DH * 2);
  _Float16* Kh   = (_Float16*)alloc((size_t)BATCH * NH * SEQ * DH * 2);
  _Float16* Vt   = (_Float16*)alloc((size_t)BATCH * NH * DH * SEQ * 2);
  _Float16* ctxh = (_Float16*)alloc((size_t)ROWS * HD * 2);
  float*    Mrow = (float*)alloc((size_t)BATCH * NH * SEQ * 4);
  float*    Lrow = (float*)alloc((size_t)BATCH * NH * SEQ * 4);

  int n = ROWS * D_IN;
  k_f32_to_f16<<<(n + 255) / 256, 256, 0, stream>>>(x, xh, n);
  k_transpose_to_f16<<<(D_IN * HD + 255) / 256, 256, 0, stream>>>(Wq, WqT, D_IN, 10);
  k_transpose_to_f16<<<(D_IN * HD + 255) / 256, 256, 0, stream>>>(Wk, WkT, D_IN, 10);
  k_transpose_to_f16<<<(D_IN * HD + 255) / 256, 256, 0, stream>>>(Wv, WvT, D_IN, 10);
  k_transpose_to_f16<<<(HD * DH + 255) / 256, 256, 0, stream>>>(Wo, WoT, HD, 6);

  k_qkv<<<3 * (ROWS / 32) * (HD / 64), 32, 0, stream>>>(xh, WqT, WkT, WvT, Qh, Kh, Vt);

  int nblk = BATCH * NH * (SEQ / 16);
  k_softmax_stats<<<nblk, 32, 0, stream>>>(Qh, Kh, Mrow, Lrow);
  k_attn_ctx<<<nblk, 32, 0, stream>>>(Qh, Kh, Vt, Mrow, Lrow, attn, ctxh);

  k_outproj<<<(ROWS / 16) * (DH / 16), 32, 0, stream>>>(ctxh, WoT, out);
}
